// TGN_78176994721831
// MI455X (gfx1250) — matrix operand order repaired and checked by reading, not compiled, not run on hardware
//
#include <hip/hip_runtime.h>
#include <hip/hip_bf16.h>

typedef __attribute__((ext_vector_type(8)))  _Float16 v8h;
typedef __attribute__((ext_vector_type(16))) _Float16 v16h;
typedef __attribute__((ext_vector_type(8)))  float    v8f;
typedef __attribute__((ext_vector_type(4)))  unsigned int v4u;
typedef __attribute__((ext_vector_type(8)))  int      v8i_;
typedef __attribute__((ext_vector_type(4)))  int      v4i;

#define K_NODES 100000
#define M_EDGES 600000
#define EMB 128
#define NHEAD 2

// ---------------------------------------------------------------------------
// float atomic max via int ordering trick (maps to global_atomic_max/min)
// ---------------------------------------------------------------------------
__device__ __forceinline__ void atomicMaxF(float* addr, float val) {
  if (val >= 0.0f) {
    atomicMax((int*)addr, __float_as_int(val));
  } else {
    atomicMin((unsigned int*)addr, __float_as_uint(val));
  }
}

// ---------------------------------------------------------------------------
// 1) Weight prep: transpose + f16-convert Wq|Wk|Wv|Wskip -> [512][256],
//    We -> [128][256], pack combined bias [512].
// ---------------------------------------------------------------------------
__global__ void tgn_prep_weights(const float* __restrict__ Wq, const float* __restrict__ bq,
                                 const float* __restrict__ Wk, const float* __restrict__ bk,
                                 const float* __restrict__ Wv, const float* __restrict__ bv,
                                 const float* __restrict__ Wskip, const float* __restrict__ bskip,
                                 const float* __restrict__ We,
                                 _Float16* __restrict__ wcomb_t,   // [512][256]
                                 float* __restrict__ bias_comb,    // [512]
                                 _Float16* __restrict__ we_t)      // [128][256]
{
  int col = blockIdx.x;           // 0..639
  int r   = threadIdx.x;          // 0..255
  if (col < 512) {
    int sub = col & 127;
    const float* W; const float* B;
    int sel = col >> 7;
    if (sel == 0)      { W = Wq;    B = bq;    }
    else if (sel == 1) { W = Wk;    B = bk;    }
    else if (sel == 2) { W = Wv;    B = bv;    }
    else               { W = Wskip; B = bskip; }
    wcomb_t[(size_t)col * 256 + r] = (_Float16)W[(size_t)r * 128 + sub];
    if (r == 0) bias_comb[col] = B[sub];
  } else {
    int sub = col - 512;          // 0..127
    we_t[(size_t)sub * 256 + r] = (_Float16)We[(size_t)r * 128 + sub];
  }
}

// ---------------------------------------------------------------------------
// 2) Init accumulators (ws is poisoned by the harness)
// ---------------------------------------------------------------------------
__global__ void tgn_init(float* __restrict__ out_accum,
                         float* __restrict__ amax,
                         float* __restrict__ denom)
{
  int i = blockIdx.x * blockDim.x + threadIdx.x;
  if (i < K_NODES * EMB) out_accum[i] = 0.0f;
  if (i < K_NODES * NHEAD) { amax[i] = -__builtin_inff(); denom[i] = 0.0f; }
}

// ---------------------------------------------------------------------------
// 3) Gather x = memory[n_id] || node_feat[n_id]  (f16), and last_update[n_id]
// ---------------------------------------------------------------------------
__global__ void tgn_gather_x(const int* __restrict__ n_id,
                             const float* __restrict__ memory,
                             const float* __restrict__ node_feat,
                             const float* __restrict__ last_update,
                             _Float16* __restrict__ x16,
                             float* __restrict__ lu_g)
{
  int k = blockIdx.x;             // 0..K-1
  int t = threadIdx.x;            // 0..255
  int nid = n_id[k];
  float v = (t < 128) ? memory[(size_t)nid * 128 + t]
                      : node_feat[(size_t)nid * 128 + (t - 128)];
  x16[(size_t)k * 256 + t] = (_Float16)v;
  if (t == 0) lu_g[k] = last_update[nid];
}

// ---------------------------------------------------------------------------
// 4) edge_attr = [ cos(rel_t*w_t + b_t) || msg ]  (f16)
// ---------------------------------------------------------------------------
__global__ void tgn_edge_attr(const int* __restrict__ ei,
                              const int* __restrict__ e_id,
                              const float* __restrict__ edge_t,
                              const float* __restrict__ lu_g,
                              const float* __restrict__ edge_raw_msg,
                              const float* __restrict__ w_t,
                              const float* __restrict__ b_t,
                              _Float16* __restrict__ ea16)
{
  int m = blockIdx.x;             // 0..M-1
  int t = threadIdx.x;            // 0..255
  int eid = e_id[m];
  float v;
  if (t < 128) {
    int src = ei[m];
    float rel = edge_t[eid] - lu_g[src];
    v = __cosf(rel * w_t[t] + b_t[t]);
  } else {
    v = edge_raw_msg[(size_t)eid * 128 + (t - 128)];
  }
  ea16[(size_t)m * 256 + t] = (_Float16)v;
}

// ---------------------------------------------------------------------------
// 5) WMMA GEMM with TDM-staged A in LDS.
//    Block = 256 threads = 8 waves, handles MT m-tiles of 16 rows.
//    TDM loads A[MT*16, 256]f16 into LDS with a 16B row pad (528B stride)
//    to spread rows across banks. Each wave computes a 16x64 N-strip:
//      wave w -> m-subtile w/NW, n_base = (w%NW)*64;  NW*64 == ncols.
//    A fragments come from LDS (ds_load_b128), B streams from global
//    (weights are L2-resident).
// ---------------------------------------------------------------------------
template<int MT, int NW>
__global__ __launch_bounds__(256)
void tgn_gemm_lds(const _Float16* __restrict__ A,
                  const _Float16* __restrict__ Bt,   // [ncols][256] f16
                  const float* __restrict__ bias,
                  float* __restrict__ C,
                  int ncols)
{
  constexpr int ROWH = 264;                      // halfs per LDS row (528B)
  __shared__ _Float16 smem[MT * 16 * ROWH];

  int row0 = blockIdx.x * (MT * 16);

#if __has_builtin(__builtin_amdgcn_tensor_load_to_lds) && __has_builtin(__builtin_amdgcn_s_wait_tensorcnt)
  if (threadIdx.x < 32) {                        // one TDM issue per block
    unsigned lds_base = (unsigned)(size_t)(void*)smem;   // low bits = LDS offset
    unsigned long long ga = (unsigned long long)(size_t)(const void*)(A + (size_t)row0 * 256);
    v4u g0;
    g0[0] = 1u;                                          // count=1 (valid), user mode
    g0[1] = lds_base;                                    // lds_addr
    g0[2] = (unsigned)(ga & 0xFFFFFFFFu);                // global_addr[31:0]
    g0[3] = (unsigned)((ga >> 32) & 0x1FFFFFFu) | (2u << 30); // addr[56:32] | type=2
    v8i_ g1;
    // data_size=4B(code 2), pad_enable, pad_interval=512B(code 6), pad_amount=4dw(code 3)
    g1[0] = (int)((2u << 16) | (1u << 20) | (6u << 22) | (3u << 25));
    g1[1] = (int)(128u << 16);                           // tensor_dim0 = 128 dwords
    g1[2] = (int)((unsigned)(MT * 16) << 16);            // tensor_dim1 = MT*16 rows
    g1[3] = (int)(128u << 16);                           // tile_dim0 = 128 dwords
    g1[4] = (int)(MT * 16);                              // tile_dim1 = MT*16
    g1[5] = 128;                                         // tensor_dim0_stride = 128 dwords
    g1[6] = 0;
    g1[7] = 0;
    v4i gz4 = {0, 0, 0, 0};
    v8i_ gz8 = {0, 0, 0, 0, 0, 0, 0, 0};
    __builtin_amdgcn_tensor_load_to_lds(g0, g1, gz4, gz4, gz8, 0);
    __builtin_amdgcn_s_wait_tensorcnt(0);
  }
#else
  for (int i = threadIdx.x; i < MT * 16 * 32; i += 256) {
    int r = i >> 5, c = i & 31;                 // 32 chunks of 8 halfs per row
    *(v8h*)&smem[r * ROWH + c * 8] = *(const v8h*)&A[(size_t)(row0 + r) * 256 + c * 8];
  }
#endif
  __syncthreads();

  int wave = threadIdx.x >> 5;
  int lane = threadIdx.x & 31;
  int msub = wave / NW;
  int n_base = (wave % NW) * 64;
  int g = lane >> 4;
  int l = lane & 15;

  v8f acc0 = {}, acc1 = {}, acc2 = {}, acc3 = {};
  const _Float16* arow = &smem[(msub * 16 + l) * ROWH];
  const _Float16* brow = Bt + (size_t)(n_base + l) * 256;

#pragma unroll
  for (int kk = 0; kk < 8; ++kk) {
    // A fragment (16-bit 16x32): lane l,g: K = 16g + j
    const _Float16* ap = arow + kk * 32 + g * 16;
    v8h alo = *(const v8h*)(ap);
    v8h ahi = *(const v8h*)(ap + 8);
    v16h a = __builtin_shufflevector(alo, ahi, 0,1,2,3,4,5,6,7,8,9,10,11,12,13,14,15);

    // B fragment (16-bit 32x16): lane l holds N=l, K = 16g + j
    const _Float16* bp = brow + kk * 32 + g * 16;
    v16h b0 = *(const v16h*)(bp);
    v16h b1 = *(const v16h*)(bp + 16 * 256);
    v16h b2 = *(const v16h*)(bp + 32 * 256);
    v16h b3 = *(const v16h*)(bp + 48 * 256);

    acc0 = __builtin_amdgcn_wmma_f32_16x16x32_f16(false, a, false, b0, (short)0, acc0, false, false);
    acc1 = __builtin_amdgcn_wmma_f32_16x16x32_f16(false, a, false, b1, (short)0, acc1, false, false);
    acc2 = __builtin_amdgcn_wmma_f32_16x16x32_f16(false, a, false, b2, (short)0, acc2, false, false);
    acc3 = __builtin_amdgcn_wmma_f32_16x16x32_f16(false, a, false, b3, (short)0, acc3, false, false);
  }

#pragma unroll
  for (int t = 0; t < 4; ++t) {
    v8f acc = (t == 0) ? acc0 : (t == 1) ? acc1 : (t == 2) ? acc2 : acc3;
    int col = n_base + t * 16 + l;
    float bv = bias ? bias[col] : 0.0f;
#pragma unroll
    for (int r = 0; r < 8; ++r) {
      // C/D layout: VGPR r -> row M = r + 8*g, lane -> column
      C[(size_t)(row0 + msub * 16 + r + 8 * g) * ncols + col] = acc[r] + bv;
    }
  }
}

// ---------------------------------------------------------------------------
// 6) Attention pass 1: per-edge alpha = <q[dst], k[src]+e>/sqrt(D), seg-max.
//    One wave per edge; lanes 0-15 = head 0, lanes 16-31 = head 1.
// ---------------------------------------------------------------------------
__global__ __launch_bounds__(256)
void tgn_attn1(const int* __restrict__ ei,
               const float* __restrict__ qkvs,   // [K,512] q|k|v|skip
               const float* __restrict__ e_f32,  // [M,128]
               float* __restrict__ alpha_out,    // [M,2]
               float* __restrict__ amax)         // [K,2]
{
  int wave = threadIdx.x >> 5;
  int lane = threadIdx.x & 31;
  int m = blockIdx.x * 8 + wave;
  if (m >= M_EDGES) return;
  int src = ei[m];
  int dst = ei[M_EDGES + m];
  int d0 = lane * 4;                                  // lanes 0-15 cover d 0..63
  const float* qrow = qkvs + (size_t)dst * 512;
  const float* krow = qkvs + (size_t)src * 512 + 128;
  const float* erow = e_f32 + (size_t)m * 128;
  float p = 0.0f;
#pragma unroll
  for (int j = 0; j < 4; ++j) {
    int d = d0 + j;
    p += qrow[d] * (krow[d] + erow[d]);
  }
  // reduce within each 16-lane half (one head each)
  for (int off = 1; off <= 8; off <<= 1) p += __shfl_xor(p, off, 32);
  if ((lane & 15) == 0) {
    int h = lane >> 4;
    float a = p * 0.125f;                             // 1/sqrt(64)
    alpha_out[(size_t)m * 2 + h] = a;
    atomicMaxF(&amax[(size_t)dst * 2 + h], a);
  }
}

// ---------------------------------------------------------------------------
// 7) Attention pass 2: w = exp(alpha - amax); denom += w;
//    out_accum[dst] += w * (v[src] + e)   (unnormalized; divide in finalize)
// ---------------------------------------------------------------------------
__global__ __launch_bounds__(256)
void tgn_attn2(const int* __restrict__ ei,
               const float* __restrict__ qkvs,
               const float* __restrict__ e_f32,
               const float* __restrict__ alpha_in,
               const float* __restrict__ amax,
               float* __restrict__ denom,        // [K,2]
               float* __restrict__ out_accum)    // [K,128]
{
  int wave = threadIdx.x >> 5;
  int lane = threadIdx.x & 31;
  int m = blockIdx.x * 8 + wave;
  if (m >= M_EDGES) return;
  int src = ei[m];
  int dst = ei[M_EDGES + m];
  int h = lane >> 4;
  float w = __expf(alpha_in[(size_t)m * 2 + h] - amax[(size_t)dst * 2 + h]);
  if ((lane & 15) == 0) atomicAdd(&denom[(size_t)dst * 2 + h], w);
  const float* vrow = qkvs + (size_t)src * 512 + 256;
  const float* erow = e_f32 + (size_t)m * 128;
  int d0 = lane * 4;
#pragma unroll
  for (int j = 0; j < 4; ++j) {
    int d = d0 + j;
    atomicAdd(&out_accum[(size_t)dst * 128 + d], w * (vrow[d] + erow[d]));
  }
}

// ---------------------------------------------------------------------------
// 8) Finalize: z = out_accum/denom + skip   (empty segments -> just skip)
// ---------------------------------------------------------------------------
__global__ void tgn_finalize(const float* __restrict__ out_accum,
                             const float* __restrict__ denom,
                             const float* __restrict__ qkvs,
                             float* __restrict__ z)
{
  int k = blockIdx.x;           // 0..K-1
  int t = threadIdx.x;          // 0..127
  int h = t >> 6;
  float dn = denom[(size_t)k * 2 + h];
  float o = (dn > 0.0f) ? out_accum[(size_t)k * 128 + t] / dn : 0.0f;
  z[(size_t)k * 128 + t] = o + qkvs[(size_t)k * 512 + 384 + t];
}

// ---------------------------------------------------------------------------
// launch
// ---------------------------------------------------------------------------
static inline size_t alignup(size_t x) { return (x + 255) & ~(size_t)255; }

extern "C" void kernel_launch(void* const* d_in, const int* in_sizes, int n_in,
                              void* d_out, int out_size, void* d_ws, size_t ws_size,
                              hipStream_t stream) {
  const int*   n_id         = (const int*)  d_in[0];
  const int*   ei           = (const int*)  d_in[1];  // [2,M] flat: src then dst
  const int*   e_id         = (const int*)  d_in[2];
  /* d_in[3] t_targets: unused by reference */
  const float* node_feat    = (const float*)d_in[4];
  const float* edge_raw_msg = (const float*)d_in[5];
  const float* edge_t       = (const float*)d_in[6];
  const float* memory       = (const float*)d_in[7];
  const float* last_update  = (const float*)d_in[8];
  const float* w_t          = (const float*)d_in[9];
  const float* b_t          = (const float*)d_in[10];
  const float* Wq           = (const float*)d_in[11];
  const float* bq           = (const float*)d_in[12];
  const float* Wk           = (const float*)d_in[13];
  const float* bk           = (const float*)d_in[14];
  const float* Wv           = (const float*)d_in[15];
  const float* bv           = (const float*)d_in[16];
  const float* We           = (const float*)d_in[17];
  const float* Wskip        = (const float*)d_in[18];
  const float* bskip        = (const float*)d_in[19];
  float* z = (float*)d_out;

  char* ws = (char*)d_ws;
  size_t off = 0;
  auto take = [&](size_t bytes) { size_t o = off; off = alignup(off + bytes); return o; };

  _Float16* wcomb_t  = (_Float16*)(ws + take((size_t)512 * 256 * 2));
  _Float16* we_t     = (_Float16*)(ws + take((size_t)128 * 256 * 2));
  float*    bias_c   = (float*)   (ws + take((size_t)512 * 4));
  _Float16* x16      = (_Float16*)(ws + take((size_t)K_NODES * 256 * 2));
  float*    lu_g     = (float*)   (ws + take((size_t)K_NODES * 4));
  _Float16* ea16     = (_Float16*)(ws + take((size_t)M_EDGES * 256 * 2));
  float*    qkvs     = (float*)   (ws + take((size_t)K_NODES * 512 * 4));
  float*    e_f32    = (float*)   (ws + take((size_t)M_EDGES * 128 * 4));
  float*    alpha    = (float*)   (ws + take((size_t)M_EDGES * 2 * 4));
  float*    amax     = (float*)   (ws + take((size_t)K_NODES * 2 * 4));
  float*    denom    = (float*)   (ws + take((size_t)K_NODES * 2 * 4));
  float*    out_acc  = (float*)   (ws + take((size_t)K_NODES * 128 * 4));
  (void)ws_size; (void)in_sizes; (void)n_in; (void)out_size;

  // 1) weights
  tgn_prep_weights<<<640, 256, 0, stream>>>(Wq, bq, Wk, bk, Wv, bv, Wskip, bskip, We,
                                            wcomb_t, bias_c, we_t);
  // 2) init accumulators
  tgn_init<<<(K_NODES * EMB + 255) / 256, 256, 0, stream>>>(out_acc, amax, denom);
  // 3) gather x
  tgn_gather_x<<<K_NODES, 256, 0, stream>>>(n_id, memory, node_feat, last_update, x16, lu_g);
  // 4) edge_attr
  tgn_edge_attr<<<M_EDGES, 256, 0, stream>>>(ei, e_id, edge_t, lu_g, edge_raw_msg,
                                             w_t, b_t, ea16);
  // 5) node GEMM: qkvs = x @ [Wq|Wk|Wv|Wskip] + bias   (K x 512)
  //    1 m-tile per block, 8 waves cover 512 cols; A staged once via TDM.
  tgn_gemm_lds<1, 8><<<K_NODES / 16, 256, 0, stream>>>(x16, wcomb_t, bias_c, qkvs, 512);
  // 5b) edge GEMM: e = edge_attr @ We                  (M x 128)
  //    4 m-tiles per block, 2 waves each cover 128 cols.
  tgn_gemm_lds<4, 2><<<M_EDGES / 64, 256, 0, stream>>>(ea16, we_t, (const float*)nullptr, e_f32, 128);
  // 6) attention pass 1 (alpha + segment max)
  tgn_attn1<<<(M_EDGES + 7) / 8, 256, 0, stream>>>(ei, qkvs, e_f32, alpha, amax);
  // 7) attention pass 2 (exp, denom, weighted scatter)
  tgn_attn2<<<(M_EDGES + 7) / 8, 256, 0, stream>>>(ei, qkvs, e_f32, alpha, amax, denom, out_acc);
  // 8) finalize: normalize + skip
  tgn_finalize<<<K_NODES, 128, 0, stream>>>(out_acc, denom, qkvs, z);
}
